// MultiheadCosinePropagationNetSimRatio_71811853189809
// MI455X (gfx1250) — matrix operand order repaired and checked by reading, not compiled, not run on hardware
//
#include <hip/hip_runtime.h>

#define NN    4096
#define DD    64
#define HH    4
#define NSTEP 4
#define INV_TEMP 5.0f
#define EPSV  1e-12f
#define NEGB  ((unsigned short)0xD9FFu)   // truncated bf16 of -9e15

typedef __attribute__((ext_vector_type(16))) __bf16 v16bf;
typedef __attribute__((ext_vector_type(8)))  __bf16 v8bf;
typedef __attribute__((ext_vector_type(8)))  float  v8f;

static __device__ __forceinline__ __bf16 f2bf_rne(float f) {  // rounding (proj only)
  unsigned u = __builtin_bit_cast(unsigned, f);
  unsigned r = (u + 0x7FFFu + ((u >> 16) & 1u)) >> 16;
  return __builtin_bit_cast(__bf16, (unsigned short)r);
}
static __device__ __forceinline__ unsigned short f2h_trunc(float f) {
  return (unsigned short)(__builtin_bit_cast(unsigned, f) >> 16);
}
static __device__ __forceinline__ float h2f(unsigned short s) {
  return __builtin_bit_cast(float, ((unsigned)s) << 16);
}
static __device__ __forceinline__ v16bf cat8(v8bf a, v8bf b) {
  return __builtin_shufflevector(a, b, 0,1,2,3,4,5,6,7,8,9,10,11,12,13,14,15);
}
static __device__ __forceinline__ v8f wmma_bf16(v16bf a, v16bf b, v8f c) {
  return __builtin_amdgcn_wmma_f32_16x16x32_bf16(false, a, false, b, (short)0, c, false, false);
}

// ---------------------------------------------------------------------------
// Stage 0: h = X @ W[h] per head, row-normalize.
// hn (bf16 [H][N][64]) : WMMA operand for the Gram matrix.
// hT (bf16 [H][64][N]) : h transposed, contiguous B-operand for P@h.
// ---------------------------------------------------------------------------
__global__ __launch_bounds__(64) void mcpn_proj(const float* __restrict__ feat,
                                                const float* __restrict__ W,
                                                __bf16* __restrict__ hn,
                                                __bf16* __restrict__ hT) {
  const int n = blockIdx.x, head = blockIdx.y, d = threadIdx.x;
  __shared__ float frow[DD];
  __shared__ float red[2];
  frow[d] = feat[n * DD + d];
  __syncthreads();
  const float* w = W + head * DD * DD;
  float acc = 0.f;
#pragma unroll
  for (int k = 0; k < DD; ++k) acc = fmaf(frow[k], w[k * DD + d], acc);
  float sq = acc * acc;
#pragma unroll
  for (int off = 16; off > 0; off >>= 1) sq += __shfl_xor(sq, off, 32);
  const int lane = threadIdx.x & 31, wave = threadIdx.x >> 5;
  if (lane == 0) red[wave] = sq;
  __syncthreads();
  const float inv = 1.f / (sqrtf(red[0] + red[1]) + EPSV);
  hn[(head * NN + n) * DD + d] = f2bf_rne(acc * inv);
  hT[(head * DD + d) * NN + n] = f2bf_rne(acc);
}

// ---------------------------------------------------------------------------
// Stage 1: fused masked-cosine attention; one head, one 16-row block.
//  P1: raw sim stripe [16 x 4096] via WMMA -> LDS bf16 (truncation stores).
//  P2: coalesced int4 adjacency stream, mask in LDS, fused row-max.
//  P3: keep-ratio threshold + exp + denominator (coalesced, b64 LDS rmw).
//  P4: out = P @ h via WMMA, A-frags from LDS (ds_load_b128), B from hT.
// grid (N/16, H), block 128 (4 waves x 1024 columns each).
// ---------------------------------------------------------------------------
#define SIM_BYTES  (16 * NN * 2)                    // 131072
#define SCR_OFF    SIM_BYTES
#define SCR_BYTES  (4 * 16 * DD * 4)                // 16384
#define PART_OFF   (SCR_OFF + SCR_BYTES)            // [16][128] f32
#define PART2_OFF  (PART_OFF + 16 * 128 * 4)        // [16][8]  f32
#define RMAX_OFF   (PART2_OFF + 16 * 8 * 4)
#define RDEN_OFF   (RMAX_OFF + 64)
#define SMEM_TOTAL (RDEN_OFF + 64)                  // 156288 bytes

__global__ __launch_bounds__(128) void mcpn_attn(const __bf16* __restrict__ hn,
                                                 const __bf16* __restrict__ hT,
                                                 const int* __restrict__ adj,
                                                 float* __restrict__ outh) {
  extern __shared__ char smem[];
  __bf16* simL    = (__bf16*)smem;
  float*  scratch = (float*)(smem + SCR_OFF);
  float*  part    = (float*)(smem + PART_OFF);
  float*  part2   = (float*)(smem + PART2_OFF);
  float*  rowmax  = (float*)(smem + RMAX_OFF);
  float*  rowden  = (float*)(smem + RDEN_OFF);

  const int head = blockIdx.y;
  const int row0 = blockIdx.x * 16;
  const int tid  = threadIdx.x;
  const int lane = tid & 31;
  const int wave = tid >> 5;
  const int lm   = lane & 15;
  const bool hi  = lane >= 16;
  const int abase = hi ? 8 : 0;    // 16-bit A-operand K-chunk base per lane half
  const int bbase = hi ? 8 : 0;    // 16-bit B-operand K-chunk base per lane half

  const __bf16* hnH = hn + (size_t)head * NN * DD;

  // A fragments: this block's 16 hn rows, K = 0..31 / 32..63
  const __bf16* arow = hnH + (size_t)(row0 + lm) * DD;
  const v16bf a0 = cat8(*(const v8bf*)(arow + abase),      *(const v8bf*)(arow + abase + 16));
  const v16bf a1 = cat8(*(const v8bf*)(arow + abase + 32), *(const v8bf*)(arow + abase + 48));

  // ---- P1: raw scaled similarity stripe -> LDS (truncation d16_hi stores) ----
  {
    unsigned short* sp0 = (unsigned short*)simL + (hi ? 8 : 0) * NN;
    for (int t = 0; t < 64; ++t) {
      const int m0 = wave * 1024 + t * 16;
      const __bf16* brow = hnH + (size_t)(m0 + lm) * DD;
      const v16bf b0 = cat8(*(const v8bf*)(brow + bbase), *(const v8bf*)(brow + bbase + 16));
      const v16bf b1 = cat8(*(const v8bf*)(brow + 32 + bbase), *(const v8bf*)(brow + 48 + bbase));
      v8f acc = {};
      acc = wmma_bf16(a0, b0, acc);
      acc = wmma_bf16(a1, b1, acc);
      unsigned short* sp = sp0 + m0 + lm;
#pragma unroll
      for (int i = 0; i < 8; ++i) sp[i * NN] = f2h_trunc(acc[i] * INV_TEMP);
    }
  }
  __syncthreads();

  // ---- P2: coalesced adjacency mask + fused row max ----
  // The block's adjacency slab adj[row0 .. row0+15][0..4095] is contiguous.
  {
    const int4* ap = (const int4*)(adj + (size_t)row0 * NN);
    ushort4* sv = (ushort4*)simL;      // 4 bf16 per slot
    float mx = -3.0e38f;
    for (int j = 0; j < 128; ++j) {    // iteration j lies entirely in row j>>3
      const int idx4 = j * 128 + tid;
      if (j < 120) __builtin_prefetch(ap + idx4 + 1024, 0, 1);
      const int4 av = ap[idx4];
      ushort4 s = sv[idx4];
      s.x = av.x > 0 ? s.x : NEGB;
      s.y = av.y > 0 ? s.y : NEGB;
      s.z = av.z > 0 ? s.z : NEGB;
      s.w = av.w > 0 ? s.w : NEGB;
      sv[idx4] = s;
      mx = fmaxf(mx, fmaxf(fmaxf(h2f(s.x), h2f(s.y)), fmaxf(h2f(s.z), h2f(s.w))));
      if ((j & 7) == 7) { part[(j >> 3) * 128 + tid] = mx; mx = -3.0e38f; }
    }
  }
  __syncthreads();
  {  // reduce [16][128] -> [16]
    const int r = tid >> 3, seg = tid & 7;
    float mx = part[r * 128 + seg * 16];
#pragma unroll
    for (int k = 1; k < 16; ++k) mx = fmaxf(mx, part[r * 128 + seg * 16 + k]);
    part2[r * 8 + seg] = mx;
  }
  __syncthreads();
  if (tid < 16) {
    float mx = part2[tid * 8];
#pragma unroll
    for (int k = 1; k < 8; ++k) mx = fmaxf(mx, part2[tid * 8 + k]);
    rowmax[tid] = mx;
  }
  __syncthreads();

  // ---- P3: keep-ratio threshold + exp + denominator ----
  {
    ushort4* sv = (ushort4*)simL;
    float sum = 0.f;
    float rm = rowmax[0], th = 0.5f * rm;
    for (int j = 0; j < 128; ++j) {
      if ((j & 7) == 0) { rm = rowmax[j >> 3]; th = 0.5f * rm; }
      const int idx4 = j * 128 + tid;
      ushort4 s = sv[idx4];
      const float v0 = h2f(s.x), v1 = h2f(s.y), v2 = h2f(s.z), v3 = h2f(s.w);
      const float p0 = (v0 >= th) ? __expf(v0 - rm) : 0.f;
      const float p1 = (v1 >= th) ? __expf(v1 - rm) : 0.f;
      const float p2 = (v2 >= th) ? __expf(v2 - rm) : 0.f;
      const float p3 = (v3 >= th) ? __expf(v3 - rm) : 0.f;
      s.x = f2h_trunc(p0); s.y = f2h_trunc(p1);
      s.z = f2h_trunc(p2); s.w = f2h_trunc(p3);
      sv[idx4] = s;
      sum += (p0 + p1) + (p2 + p3);
      if ((j & 7) == 7) { part[(j >> 3) * 128 + tid] = sum; sum = 0.f; }
    }
  }
  __syncthreads();
  {
    const int r = tid >> 3, seg = tid & 7;
    float s = 0.f;
#pragma unroll
    for (int k = 0; k < 16; ++k) s += part[r * 128 + seg * 16 + k];
    part2[r * 8 + seg] = s;
  }
  __syncthreads();
  if (tid < 16) {
    float s = 0.f;
#pragma unroll
    for (int k = 0; k < 8; ++k) s += part2[tid * 8 + k];
    rowden[tid] = s;
  }
  __syncthreads();

  // ---- P4: out = P @ h; A-frags from LDS, B-frags contiguous from hT ----
  v8f acc2[4] = {v8f{}, v8f{}, v8f{}, v8f{}};
  const __bf16* hTh = hT + (size_t)head * DD * NN;
  for (int kt = 0; kt < 32; ++kt) {
    const int kb = wave * 1024 + kt * 32;
    const __bf16* prow = simL + lm * NN + kb + abase;
    const v16bf pa = cat8(*(const v8bf*)prow, *(const v8bf*)(prow + 16));
#pragma unroll
    for (int dt = 0; dt < 4; ++dt) {
      const __bf16* hp = hTh + (size_t)(dt * 16 + lm) * NN + kb + bbase;
      const v16bf hb = cat8(*(const v8bf*)hp, *(const v8bf*)(hp + 16));
      acc2[dt] = wmma_bf16(pa, hb, acc2[dt]);
    }
  }
#pragma unroll
  for (int dt = 0; dt < 4; ++dt)
#pragma unroll
    for (int i = 0; i < 8; ++i) {
      const int r = i + (hi ? 8 : 0);
      scratch[wave * (16 * DD) + r * DD + dt * 16 + lm] = acc2[dt][i];
    }
  __syncthreads();

  for (int e = tid; e < 16 * DD; e += 128) {
    const int r = e >> 6, d = e & 63;
    float s = scratch[r * DD + d] + scratch[1024 + r * DD + d] +
              scratch[2048 + r * DD + d] + scratch[3072 + r * DD + d];
    const float den = rowden[r];
    const float att = (den > 0.f) ? (s / den) : 0.f;
    const float o = (att > 0.f) ? att : (__expf(att) - 1.f);   // elu
    outh[((size_t)head * NN + row0 + r) * DD + d] = o;
  }
}

// ---------------------------------------------------------------------------
// Stage 2: mean over heads (elu already applied per head).
// ---------------------------------------------------------------------------
__global__ __launch_bounds__(256) void mcpn_combine(const float* __restrict__ outh,
                                                    float* __restrict__ dst) {
  const int idx = blockIdx.x * 256 + threadIdx.x;
  float s = 0.f;
#pragma unroll
  for (int h = 0; h < HH; ++h) s += outh[(size_t)h * NN * DD + idx];
  dst[idx] = 0.25f * s;
}

// ---------------------------------------------------------------------------
extern "C" void kernel_launch(void* const* d_in, const int* in_sizes, int n_in,
                              void* d_out, int out_size, void* d_ws, size_t ws_size,
                              hipStream_t stream) {
  (void)in_sizes; (void)n_in; (void)out_size; (void)ws_size;
  const float* feat0 = (const float*)d_in[0];
  const int*   adj   = (const int*)d_in[1];
  const float* Ws    = (const float*)d_in[2];
  float* outp = (float*)d_out;

  char* w = (char*)d_ws;
  __bf16* hn    = (__bf16*)w;  w += (size_t)HH * NN * DD * 2;   // 2 MB
  __bf16* hT    = (__bf16*)w;  w += (size_t)HH * DD * NN * 2;   // 2 MB
  float*  outh  = (float*)w;   w += (size_t)HH * NN * DD * 4;   // 4 MB
  float*  featb = (float*)w;                                     // 1 MB

  hipFuncSetAttribute(reinterpret_cast<const void*>(mcpn_attn),
                      hipFuncAttributeMaxDynamicSharedMemorySize, SMEM_TOTAL);

  for (int s = 0; s < NSTEP; ++s) {
    const float* f = (s == 0) ? feat0 : featb;
    mcpn_proj<<<dim3(NN, HH), 64, 0, stream>>>(f, Ws + (size_t)s * HH * DD * DD, hn, hT);
    mcpn_attn<<<dim3(NN / 16, HH), 128, SMEM_TOTAL, stream>>>(
        hn, hT, adj + (size_t)s * NN * NN, outh);
    float* dst = (s == NSTEP - 1) ? outp : featb;
    mcpn_combine<<<(NN * DD) / 256, 256, 0, stream>>>(outh, dst);
  }
}
